// CBERT_24257975288162
// MI455X (gfx1250) — matrix operand order repaired and checked by reading, not compiled, not run on hardware
//
#include <hip/hip_runtime.h>
#include <math.h>

// ---------------- CDNA5 WMMA / TDM types ----------------
typedef __bf16 bhalf;
typedef __attribute__((ext_vector_type(16))) bhalf v16bf;
typedef __attribute__((ext_vector_type(8)))  float v8f;
typedef __attribute__((ext_vector_type(4)))  unsigned int u32x4;
typedef __attribute__((ext_vector_type(8)))  int i32x8;
typedef __attribute__((ext_vector_type(4)))  int i32x4;

// ---------------- model dims ----------------
namespace {
constexpr int kB = 2, kLT = 128, kLC = 256;
constexpr int kD = 768, kI = 3072, kH = 12, kHD = 64, kE = 8, kNL = 2;
constexpr int kCLFIN = 2 * kD + 4;   // 1540
constexpr int kNLAB = 8;
}

// f32 -> bf16 bits, round-to-nearest-even
__device__ __forceinline__ unsigned short f2bf(float f) {
  union { float f; unsigned int u; } v; v.f = f;
  unsigned int u = v.u;
  unsigned int r = (u + 0x7FFFu + ((u >> 16) & 1u)) >> 16;
  return (unsigned short)r;
}
__device__ __forceinline__ unsigned int pack2(float a, float b) {
  return (unsigned int)f2bf(a) | ((unsigned int)f2bf(b) << 16);
}

// =====================================================================
// Double-buffered WMMA GEMM: C[MxN] = act(A[MxK] * W[KxN] + bias) (f32 io)
// Block tile 64x128, K-step 32, 256 threads = 8 waves; each wave: 4x
// v_wmma_f32_16x16x32_bf16 accumulators. Global loads for tile k+1 are
// issued before computing tile k (ping-pong LDS), hiding HBM/L2 latency.
// mode 0: store; mode 1: C[row,col] += gate[row*gstride]*val (expert accum)
// act 0: none; act 1: exact GELU.  Requires M%64==0, N%128==0, K%32==0.
// =====================================================================
#define GBM 64
#define GBN 128
#define GBK 32
#define GPAD 17   // uints per LDS row (16 bf16 K-pairs + 1 pad)

__global__ __launch_bounds__(256)
void wmma_gemm_kernel(const float* __restrict__ A, const float* __restrict__ W,
                      const float* __restrict__ bias, float* __restrict__ C,
                      int M, int N, int K, int act, int mode,
                      const float* __restrict__ gate, int gstride)
{
  __shared__ unsigned int Al[2][GBM][GPAD];   // A tiles: row-major bf16 pairs
  __shared__ unsigned int Bl[2][GBN][GPAD];   // W tiles: column-major bf16 pairs

  const int tid  = threadIdx.x;
  const int lane = tid & 31;
  const int wid  = tid >> 5;
  const int wm   = wid & 3;      // 16-row block within 64
  const int wn   = wid >> 2;     // 64-col block within 128
  const int m0   = blockIdx.y * GBM;
  const int n0   = blockIdx.x * GBN;
  const int mrow = lane & 15;
  const int hi   = lane >> 4;

  v8f acc[4];
  { v8f z = {0.f,0.f,0.f,0.f,0.f,0.f,0.f,0.f}; acc[0]=z; acc[1]=z; acc[2]=z; acc[3]=z; }

  const int ar = tid >> 3;          // A loader row 0..31
  const int ak = (tid & 7) << 2;    // A loader K offset (float4)

  float4 fa[2], fw[4];
  auto gload = [&](int k0) {
#pragma unroll
    for (int it = 0; it < 2; ++it)
      fa[it] = *(const float4*)(A + (size_t)(m0 + ar + it * 32) * K + k0 + ak);
#pragma unroll
    for (int it = 0; it < 4; ++it) {
      const int qq = it * 256 + tid;
      fw[it] = *(const float4*)(W + (size_t)(k0 + (qq >> 5)) * N + n0 + ((qq & 31) << 2));
    }
  };
  auto sstore = [&](int pb) {
#pragma unroll
    for (int it = 0; it < 2; ++it) {
      const int r = ar + it * 32;
      Al[pb][r][(ak >> 1) + 0] = pack2(fa[it].x, fa[it].y);
      Al[pb][r][(ak >> 1) + 1] = pack2(fa[it].z, fa[it].w);
    }
#pragma unroll
    for (int it = 0; it < 4; ++it) {
      const int qq = it * 256 + tid;
      const int kk = qq >> 5;
      const int nq = (qq & 31) << 2;
      ((unsigned short*)Bl[pb][nq + 0])[kk] = f2bf(fw[it].x);
      ((unsigned short*)Bl[pb][nq + 1])[kk] = f2bf(fw[it].y);
      ((unsigned short*)Bl[pb][nq + 2])[kk] = f2bf(fw[it].z);
      ((unsigned short*)Bl[pb][nq + 3])[kk] = f2bf(fw[it].w);
    }
  };

  gload(0);
  sstore(0);
  __syncthreads();

  int pb = 0;
  for (int k0 = 0; k0 < K; k0 += GBK) {
    const bool nxt = (k0 + GBK) < K;
    if (nxt) gload(k0 + GBK);                 // next tile's loads in flight
    if (k0 + 2 * GBK < K) {                   // L2 prefetch two tiles ahead
      __builtin_prefetch(A + (size_t)(m0 + ar) * K + k0 + 2 * GBK + ak, 0, 0);
      __builtin_prefetch(W + (size_t)(k0 + 2 * GBK + (tid >> 5)) * N + n0 + ((tid & 31) << 2), 0, 0);
    }

    // A fragment (contiguous K: element s holds k = 16*hi + s, same as B)
    union { unsigned int u[8]; v16bf v; } af;
    {
      const unsigned int* arow = Al[pb][wm * 16 + mrow];
      const int kb = hi << 3;                 // lanes 16-31 hold K=16..31
#pragma unroll
      for (int j = 0; j < 4; ++j) { af.u[j] = arow[kb + j]; af.u[4 + j] = arow[kb + 4 + j]; }
    }
    // 4 B fragments + WMMA
#pragma unroll
    for (int t = 0; t < 4; ++t) {
      union { unsigned int u[8]; v16bf v; } bf_;
      const unsigned int* bcol = Bl[pb][wn * 64 + t * 16 + mrow];
      const int kb = hi << 3;                 // lanes 16-31 hold K=16..31
#pragma unroll
      for (int j = 0; j < 8; ++j) bf_.u[j] = bcol[kb + j];
      acc[t] = __builtin_amdgcn_wmma_f32_16x16x32_bf16(
          false, af.v, false, bf_.v, (short)0, acc[t], false, false);
    }
    if (nxt) sstore(pb ^ 1);                  // convert + stage into other buffer
    __syncthreads();
    pb ^= 1;
  }

  // epilogue: bias / activation / (gated accumulate | store)
#pragma unroll
  for (int t = 0; t < 4; ++t) {
    const int col = n0 + wn * 64 + t * 16 + mrow;
    const float bv = bias ? bias[col] : 0.f;
#pragma unroll
    for (int r = 0; r < 8; ++r) {
      const int row = m0 + wm * 16 + r + (hi << 3);
      float val = acc[t][r] + bv;
      if (act == 1) val = 0.5f * val * (1.0f + erff(val * 0.7071067811865475f));
      float* dst = C + (size_t)row * N + col;
      if (mode == 1) *dst += gate[(size_t)row * gstride] * val;
      else           *dst  = val;
    }
  }
}

// =====================================================================
// Embedding: out[b,l,:] = word[id]+pos[l](+type[tid]); rows 0,1 := word[0]
// =====================================================================
__global__ __launch_bounds__(256)
void embed_kernel(const int* __restrict__ ids, const int* __restrict__ type_ids,
                  const float* __restrict__ word, const float* __restrict__ pos,
                  const float* __restrict__ type_tab, float* __restrict__ out, int L)
{
  const int token = blockIdx.x;
  const int l = token % L;
  for (int d = threadIdx.x; d < kD; d += 256) {
    float v;
    if (l < 2) {
      v = word[d];
    } else {
      v = word[(size_t)ids[token] * kD + d] + pos[(size_t)l * kD + d];
      if (type_tab) v += type_tab[(size_t)type_ids[token] * kD + d];
    }
    out[(size_t)token * kD + d] = v;
  }
}

// cross-CLS query patch: q[b,1,:] = cross_hid[b,0,:] @ Wq + bq
__global__ __launch_bounds__(256)
void crossq_kernel(const float* __restrict__ cross_hid, int Lc,
                   const float* __restrict__ Wq, const float* __restrict__ bq,
                   float* __restrict__ q, int L)
{
  const int b = blockIdx.y;
  const int j = blockIdx.x * 256 + threadIdx.x;
  const float* c = cross_hid + (size_t)b * Lc * kD;   // row 0
  float s = bq[j];
  for (int d = 0; d < kD; ++d) s += c[d] * Wq[(size_t)d * kD + j];
  q[((size_t)b * L + 1) * kD + j] = s;
}

// =====================================================================
// Attention, one block per (b,h). K/V head slices (L x 64 f32, row stride
// 768 f32) staged into LDS by the Tensor Data Mover (tensor_load_to_lds,
// issued by wave 0, completion via s_wait_tensorcnt), then all L query
// rows are processed block-cooperatively from LDS.
// =====================================================================
__global__ __launch_bounds__(256)
void attn_kernel(const float* __restrict__ q, const float* __restrict__ k,
                 const float* __restrict__ v, const int* __restrict__ mask,
                 float* __restrict__ ctx, int L)
{
  const int bh = blockIdx.x;
  const int b = bh / kH, h = bh % kH;
  const int tid = threadIdx.x;
  __shared__ float Ksh[kLC * kHD];   // 64 KB
  __shared__ float Vsh[kLC * kHD];   // 64 KB
  __shared__ float qs[kHD];
  __shared__ float sc[256];
  __shared__ float red[256];
  __shared__ float mb[256];

  const size_t base = (size_t)b * L * kD + (size_t)h * kHD;

  if (tid < 32) {   // wave 0 issues the two TDM descriptors
    const unsigned long long gak = (unsigned long long)(size_t)(const void*)(k + base);
    const unsigned long long gav = (unsigned long long)(size_t)(const void*)(v + base);
    const unsigned ldk = (unsigned)(size_t)(void*)&Ksh[0];
    const unsigned ldv = (unsigned)(size_t)(void*)&Vsh[0];
    i32x8 g1;
    g1[0] = (2 << 16);                                // mask=0 | data_size=4B
    g1[1] = (int)(((unsigned)kHD & 0xFFFFu) << 16);   // tensor_dim0[15:0] = 64
    g1[2] = (int)(((unsigned)L & 0xFFFFu) << 16);     // dim0 hi=0 | tensor_dim1[15:0] = L
    g1[3] = (int)(((unsigned)kHD) << 16);             // dim1 hi=0 | tile_dim0 = 64
    g1[4] = (int)(L & 0xFFFF);                        // tile_dim1 = L | tile_dim2 = 0
    g1[5] = kD;                                       // tensor_dim0_stride = 768
    g1[6] = 0; g1[7] = 0;                             // dim1_stride = 0 (2D)
    const i32x4 z4 = {0, 0, 0, 0};
    const i32x8 z8 = {0, 0, 0, 0, 0, 0, 0, 0};
    u32x4 g0;
    g0.x = 1u;                                        // count=1, user mode
    g0.y = ldk;                                       // lds_addr
    g0.z = (unsigned)(gak & 0xFFFFFFFFull);           // global_addr lo
    g0.w = (unsigned)((gak >> 32) & 0x1FFFFFFull) | (2u << 30);  // hi | type=2
    __builtin_amdgcn_tensor_load_to_lds(g0, g1, z4, z4, z8, 0);
    g0.y = ldv;
    g0.z = (unsigned)(gav & 0xFFFFFFFFull);
    g0.w = (unsigned)((gav >> 32) & 0x1FFFFFFull) | (2u << 30);
    __builtin_amdgcn_tensor_load_to_lds(g0, g1, z4, z4, z8, 0);
    __builtin_amdgcn_s_wait_tensorcnt(0);
  }
  if (tid < L) mb[tid] = (1.0f - (float)mask[b * L + tid]) * (-1.0e4f);
  __syncthreads();

  for (int qi = 0; qi < L; ++qi) {
    if (tid < kHD) qs[tid] = q[(size_t)b * L * kD + (size_t)qi * kD + h * kHD + tid];
    __syncthreads();
    if (tid < L) {
      const float* kv = &Ksh[tid * kHD];
      float s = 0.f;
#pragma unroll 8
      for (int d = 0; d < kHD; ++d) s += qs[d] * kv[d];
      sc[tid] = s * 0.125f + mb[tid];
    } else {
      sc[tid] = -3.0e38f;
    }
    __syncthreads();
    red[tid] = sc[tid];
    __syncthreads();
    for (int st = 128; st > 0; st >>= 1) { if (tid < st) red[tid] = fmaxf(red[tid], red[tid + st]); __syncthreads(); }
    const float mx = red[0];
    __syncthreads();
    const float e = (tid < L) ? expf(sc[tid] - mx) : 0.f;
    sc[tid] = e;
    red[tid] = e;
    __syncthreads();
    for (int st = 128; st > 0; st >>= 1) { if (tid < st) red[tid] += red[tid + st]; __syncthreads(); }
    const float inv = 1.0f / red[0];
    __syncthreads();
    if (tid < kHD) {
      float a = 0.f;
      for (int kk = 0; kk < L; ++kk) a += sc[kk] * Vsh[kk * kHD + tid];
      ctx[(size_t)b * L * kD + (size_t)qi * kD + h * kHD + tid] = a * inv;
    }
    __syncthreads();
  }
}

// x = LN(x + y) * g + b   (one token per block, D=768 = 256*3)
__global__ __launch_bounds__(256)
void add_ln_kernel(float* __restrict__ x, const float* __restrict__ y,
                   const float* __restrict__ g, const float* __restrict__ bb)
{
  const int t = blockIdx.x;
  const int tid = threadIdx.x;
  __shared__ float red[256];
  float s[3]; float lsum = 0.f;
#pragma unroll
  for (int j = 0; j < 3; ++j) {
    const int d = tid + j * 256;
    const float val = x[(size_t)t * kD + d] + y[(size_t)t * kD + d];
    s[j] = val; lsum += val;
  }
  red[tid] = lsum; __syncthreads();
  for (int st = 128; st > 0; st >>= 1) { if (tid < st) red[tid] += red[tid + st]; __syncthreads(); }
  const float mu = red[0] * (1.0f / kD);
  __syncthreads();
  float lsq = 0.f;
#pragma unroll
  for (int j = 0; j < 3; ++j) { const float d0 = s[j] - mu; lsq += d0 * d0; }
  red[tid] = lsq; __syncthreads();
  for (int st = 128; st > 0; st >>= 1) { if (tid < st) red[tid] += red[tid + st]; __syncthreads(); }
  const float rs = rsqrtf(red[0] * (1.0f / kD) + 1e-12f);
  __syncthreads();
#pragma unroll
  for (int j = 0; j < 3; ++j) {
    const int d = tid + j * 256;
    x[(size_t)t * kD + d] = (s[j] - mu) * rs * g[d] + bb[d];
  }
}

// top-2-of-8 gating: w_full[t,:]  (softmax -> top2 -> renormalize)
__global__ __launch_bounds__(256)
void gate_kernel(const float* __restrict__ x, const float* __restrict__ gW,
                 const float* __restrict__ gb, float* __restrict__ wfull)
{
  const int t = blockIdx.x;
  const int tid = threadIdx.x;
  __shared__ float acc[kE];
  if (tid < kE) acc[tid] = 0.f;
  __syncthreads();
  float p[kE] = {0.f,0.f,0.f,0.f,0.f,0.f,0.f,0.f};
  for (int d = tid; d < kD; d += 256) {
    const float xv = x[(size_t)t * kD + d];
#pragma unroll
    for (int e = 0; e < kE; ++e) p[e] += xv * gW[(size_t)d * kE + e];
  }
#pragma unroll
  for (int e = 0; e < kE; ++e) atomicAdd(&acc[e], p[e]);
  __syncthreads();
  if (tid == 0) {
    float lg[kE]; float mx = -3.0e38f;
#pragma unroll
    for (int e = 0; e < kE; ++e) { lg[e] = acc[e] + gb[e]; mx = fmaxf(mx, lg[e]); }
    float se = 0.f;
#pragma unroll
    for (int e = 0; e < kE; ++e) { lg[e] = expf(lg[e] - mx); se += lg[e]; }
    const float inv = 1.0f / se;
#pragma unroll
    for (int e = 0; e < kE; ++e) lg[e] *= inv;
    int i1 = 0;
    for (int e = 1; e < kE; ++e) if (lg[e] > lg[i1]) i1 = e;
    int i2 = (i1 == 0) ? 1 : 0;
    for (int e = 0; e < kE; ++e) if (e != i1 && lg[e] > lg[i2]) i2 = e;
    const float wsum = lg[i1] + lg[i2];
#pragma unroll
    for (int e = 0; e < kE; ++e) wfull[(size_t)t * kE + e] = 0.f;
    wfull[(size_t)t * kE + i1] = lg[i1] / wsum;
    wfull[(size_t)t * kE + i2] = lg[i2] / wsum;
  }
}

__global__ __launch_bounds__(256)
void zero_kernel(float* __restrict__ p, int n)
{
  const int i = blockIdx.x * 256 + threadIdx.x;
  if (i < n) p[i] = 0.f;
}

// classifier head
__global__ __launch_bounds__(256)
void feat_kernel(const float* __restrict__ t_hid, const float* __restrict__ c_hid,
                 const float* __restrict__ ef, float* __restrict__ feat)
{
  const int b = blockIdx.x;
  for (int d = threadIdx.x; d < kD; d += 256) {
    feat[(size_t)b * kCLFIN + d]      = t_hid[(size_t)b * kLT * kD + d];
    feat[(size_t)b * kCLFIN + kD + d] = c_hid[(size_t)b * kLC * kD + d];
  }
  if (threadIdx.x < 4) feat[(size_t)b * kCLFIN + 2 * kD + threadIdx.x] = ef[b * 4 + threadIdx.x];
}

__global__ __launch_bounds__(256)
void clf1_kernel(const float* __restrict__ feat, const float* __restrict__ W1,
                 const float* __restrict__ b1, float* __restrict__ hbuf)
{
  const int b = blockIdx.y;
  const int j = blockIdx.x * 256 + threadIdx.x;
  const float* f = feat + (size_t)b * kCLFIN;
  float s = b1[j];
  for (int d = 0; d < kCLFIN; ++d) s += f[d] * W1[(size_t)d * kD + j];
  hbuf[(size_t)b * kD + j] = fmaxf(s, 0.f);
}

__global__ __launch_bounds__(32)
void clf2_kernel(const float* __restrict__ hbuf, const float* __restrict__ W2,
                 const float* __restrict__ b2, float* __restrict__ out)
{
  const int idx = threadIdx.x;
  if (idx >= kB * kNLAB) return;
  const int b = idx >> 3, j = idx & 7;
  float s = b2[j];
  for (int d = 0; d < kD; ++d) s += hbuf[(size_t)b * kD + d] * W2[(size_t)d * kNLAB + j];
  out[b * kNLAB + j] = s;
}

// ===================== host side =====================
namespace {

struct TowerP {
  const float *W1,*W2,*Wk,*Wo,*Wq,*Wv,*b1,*b2,*bk,*bo,*bq,*bv,
              *gW,*gb,*ln1b,*ln1g,*ln2b,*ln2g,*pos,*type,*word;
};
struct Ws {
  float *t_hid,*c_hid,*q,*k,*v,*ctx,*tmp,*moe_h,*moe_out,*gatew,*feat,*clfh;
};

inline void launch_gemm(hipStream_t s, const float* A, const float* W, const float* bias,
                        float* C, int M, int N, int K, int act, int mode,
                        const float* gate, int gstride)
{
  dim3 grid(N / GBN, M / GBM);
  wmma_gemm_kernel<<<grid, 256, 0, s>>>(A, W, bias, C, M, N, K, act, mode, gate, gstride);
}

inline void run_layer(hipStream_t s, float* hid, int L, const float* cross_hid, int Lcross,
                      const int* mask, const TowerP& P, int li, const Ws& w)
{
  const int T = kB * L;
  const float* Wq = P.Wq + (size_t)li * kD * kD;  const float* bq = P.bq + (size_t)li * kD;
  const float* Wk = P.Wk + (size_t)li * kD * kD;  const float* bk = P.bk + (size_t)li * kD;
  const float* Wv = P.Wv + (size_t)li * kD * kD;  const float* bv = P.bv + (size_t)li * kD;
  const float* Wo = P.Wo + (size_t)li * kD * kD;  const float* bo = P.bo + (size_t)li * kD;

  launch_gemm(s, hid, Wq, bq, w.q, T, kD, kD, 0, 0, nullptr, 0);
  crossq_kernel<<<dim3(kD / 256, kB), 256, 0, s>>>(cross_hid, Lcross, Wq, bq, w.q, L);
  launch_gemm(s, hid, Wk, bk, w.k, T, kD, kD, 0, 0, nullptr, 0);
  launch_gemm(s, hid, Wv, bv, w.v, T, kD, kD, 0, 0, nullptr, 0);
  attn_kernel<<<dim3(kB * kH), 256, 0, s>>>(w.q, w.k, w.v, mask, w.ctx, L);
  launch_gemm(s, w.ctx, Wo, bo, w.tmp, T, kD, kD, 0, 0, nullptr, 0);
  add_ln_kernel<<<T, 256, 0, s>>>(hid, w.tmp, P.ln1g + (size_t)li * kD, P.ln1b + (size_t)li * kD);

  // MoE: gate, then dense per-expert GEMM pairs with gated accumulation
  gate_kernel<<<T, 256, 0, s>>>(hid, P.gW + (size_t)li * kD * kE, P.gb + (size_t)li * kE, w.gatew);
  zero_kernel<<<(T * kD + 255) / 256, 256, 0, s>>>(w.moe_out, T * kD);
  for (int e = 0; e < kE; ++e) {
    const size_t we = (size_t)(li * kE + e);
    launch_gemm(s, hid, P.W1 + we * kD * kI, P.b1 + we * kI, w.moe_h,
                T, kI, kD, /*gelu*/1, 0, nullptr, 0);
    launch_gemm(s, w.moe_h, P.W2 + we * kI * kD, P.b2 + we * kD, w.moe_out,
                T, kD, kI, 0, /*gated accum*/1, w.gatew + e, kE);
  }
  add_ln_kernel<<<T, 256, 0, s>>>(hid, w.moe_out, P.ln2g + (size_t)li * kD, P.ln2b + (size_t)li * kD);
}

} // namespace

extern "C" void kernel_launch(void* const* d_in, const int* in_sizes, int n_in,
                              void* d_out, int out_size, void* d_ws, size_t ws_size,
                              hipStream_t stream)
{
  (void)in_sizes; (void)n_in; (void)out_size; (void)ws_size;
  const int* text_ids  = (const int*)d_in[0];
  const int* text_mask = (const int*)d_in[1];
  const int* code_ids  = (const int*)d_in[2];
  const int* code_type = (const int*)d_in[3];
  const int* code_mask = (const int*)d_in[4];
  const float* ef      = (const float*)d_in[5];
  auto F = [&](int i) { return (const float*)d_in[i]; };
  // nested dicts flatten in sorted-key (pytree) order:
  // W1 W2 Wk Wo Wq Wv b1 b2 bk bo bq bv gW gb ln1b ln1g ln2b ln2g pos [type] word
  TowerP tp { F(6),F(7),F(8),F(9),F(10),F(11),F(12),F(13),F(14),F(15),F(16),F(17),
              F(18),F(19),F(20),F(21),F(22),F(23),F(24),nullptr,F(25) };
  TowerP cp { F(26),F(27),F(28),F(29),F(30),F(31),F(32),F(33),F(34),F(35),F(36),F(37),
              F(38),F(39),F(40),F(41),F(42),F(43),F(44),F(45),F(46) };
  const float* clfW1 = F(47); const float* clfW2 = F(48);
  const float* clfb1 = F(49); const float* clfb2 = F(50);

  // workspace carve-out (floats)
  float* ws = (float*)d_ws;
  Ws w;
  w.t_hid   = ws;
  w.c_hid   = w.t_hid   + kB * kLT * kD;
  w.q       = w.c_hid   + kB * kLC * kD;
  w.k       = w.q       + kB * kLC * kD;
  w.v       = w.k       + kB * kLC * kD;
  w.ctx     = w.v       + kB * kLC * kD;
  w.tmp     = w.ctx     + kB * kLC * kD;
  w.moe_h   = w.tmp     + kB * kLC * kD;   // 512*3072
  w.moe_out = w.moe_h   + kB * kLC * kI;
  w.gatew   = w.moe_out + kB * kLC * kD;
  w.feat    = w.gatew   + kB * kLC * kE;
  w.clfh    = w.feat    + kB * kCLFIN;

  // embeddings (rows 0,1 overwritten with word[0])
  embed_kernel<<<kB * kLT, 256, 0, stream>>>(text_ids, text_ids, tp.word, tp.pos, nullptr, w.t_hid, kLT);
  embed_kernel<<<kB * kLC, 256, 0, stream>>>(code_ids, code_type, cp.word, cp.pos, cp.type, w.c_hid, kLC);

  // cross-tower layers (text uses code CLS, then code uses fresh text CLS)
  for (int i = 0; i < kNL; ++i) {
    run_layer(stream, w.t_hid, kLT, w.c_hid, kLC, text_mask, tp, i, w);
    run_layer(stream, w.c_hid, kLC, w.t_hid, kLT, code_mask, cp, i, w);
  }

  // classifier head
  feat_kernel<<<kB, 256, 0, stream>>>(w.t_hid, w.c_hid, ef, w.feat);
  clf1_kernel<<<dim3(kD / 256, kB), 256, 0, stream>>>(w.feat, clfW1, clfb1, w.clfh);
  clf2_kernel<<<1, 32, 0, stream>>>(w.clfh, clfW2, clfb2, (float*)d_out);
}
